// GraphAttentionLayer_69793218560246
// MI455X (gfx1250) — matrix operand order repaired and checked by reading, not compiled, not run on hardware
//
#include <hip/hip_runtime.h>
#include <hip/hip_bf16.h>

typedef __attribute__((ext_vector_type(2)))  float   v2f;
typedef __attribute__((ext_vector_type(8)))  float   v8f;
typedef __attribute__((ext_vector_type(16))) __bf16  v16bf;

#define N_NODES 4096
#define IN_F    256
#define NH      4
#define HD      64
#define ALPHA   0.2f

// ---------------------------------------------------------------------------
// Kernel 1: h[N,256] = x[N,256] @ W[256,256]^T  using fp32 WMMA 16x16x4.
// One wave per 16x16 output tile. A: 16x4 f32 (2 VGPR), B: 4x16 f32 (2 VGPR).
// A layout: lane m=l&15; half=l>>4 selects K pair {0,1} vs {2,3}.
// ---------------------------------------------------------------------------
__global__ void gat_gemm_h(const float* __restrict__ x,
                           const float* __restrict__ W,
                           float* __restrict__ h) {
  int wave = (blockIdx.x * blockDim.x + threadIdx.x) >> 5;  // 0..4095
  int lane = threadIdx.x & 31;
  int tm = wave >> 4;          // 0..255 row tile
  int tn = wave & 15;          // 0..15 col tile
  int m    = lane & 15;
  int half = lane >> 4;
  const float* xrow = x + (size_t)(tm * 16 + m) * IN_F;   // A row (node)
  const float* wrow = W + (size_t)(tn * 16 + m) * IN_F;   // B col source (W row)
  v8f c = {};
  for (int kc = 0; kc < IN_F; kc += 4) {
    int k = kc + half * 2;
    v2f a = *(const v2f*)(xrow + k);
    v2f b = *(const v2f*)(wrow + k);
    c = __builtin_amdgcn_wmma_f32_16x16x4_f32(false, a, false, b,
                                              (short)0, c, false, false);
  }
  // C layout: VGPR r -> row = r + half*8, col = lane&15
  int ncol = tn * 16 + m;
#pragma unroll
  for (int r = 0; r < 8; ++r) {
    h[(size_t)(tm * 16 + r + half * 8) * IN_F + ncol] = c[r];
  }
}

// ---------------------------------------------------------------------------
// Kernel 2: e1[h][n], e2[h][n] attention dot products. One thread per (n,head).
// ---------------------------------------------------------------------------
__global__ void gat_scores(const float* __restrict__ h,
                           const float* __restrict__ att_src,
                           const float* __restrict__ att_dst,
                           float* __restrict__ e1, float* __restrict__ e2) {
  int t = blockIdx.x * blockDim.x + threadIdx.x;
  if (t >= N_NODES * NH) return;
  int n = t >> 2, hd = t & 3;
  const float* hp = h + (size_t)n * IN_F + hd * HD;
  const float* as = att_src + hd * HD;
  const float* ad = att_dst + hd * HD;
  float s1 = 0.f, s2 = 0.f;
#pragma unroll 4
  for (int d = 0; d < HD; ++d) { float v = hp[d]; s1 += v * as[d]; s2 += v * ad[d]; }
  e1[hd * N_NODES + n] = s1;
  e2[hd * N_NODES + n] = s2;
}

// ---------------------------------------------------------------------------
// Kernel 3: transpose h -> hbT[c][n] in bf16 (c = head*64+d). V operand store.
// ---------------------------------------------------------------------------
__global__ void gat_transpose_bf16(const float* __restrict__ h,
                                   __bf16* __restrict__ hbT) {
  int t = blockIdx.x * blockDim.x + threadIdx.x;   // over 4096*256
  int n = t & (N_NODES - 1);
  int c = t >> 12;
  hbT[(size_t)c * N_NODES + n] = (__bf16)h[(size_t)n * IN_F + c];
}

// ---------------------------------------------------------------------------
// Kernel 4: pack adj (int32 0/1) into 32-bit masks with wave32 ballot.
// ---------------------------------------------------------------------------
__global__ void gat_pack_adj(const int* __restrict__ adj,
                             unsigned* __restrict__ bits) {
  size_t t = (size_t)blockIdx.x * blockDim.x + threadIdx.x;  // 16M threads
  int v = adj[t];
  unsigned long long mb = __ballot(v > 0);
  if ((threadIdx.x & 31) == 0) bits[t >> 5] = (unsigned)mb;
}

// ---------------------------------------------------------------------------
// Kernel 5: per-head max of e2 (softmax upper-bound ingredient).
// ---------------------------------------------------------------------------
__global__ void gat_max_e2(const float* __restrict__ e2, float* __restrict__ m2) {
  __shared__ float red[256];
  float mx = -1e30f;
  for (int i = threadIdx.x; i < N_NODES; i += 256)
    mx = fmaxf(mx, e2[blockIdx.x * N_NODES + i]);
  red[threadIdx.x] = mx;
  __syncthreads();
  for (int s = 128; s > 0; s >>= 1) {
    if ((int)threadIdx.x < s) red[threadIdx.x] = fmaxf(red[threadIdx.x], red[threadIdx.x + s]);
    __syncthreads();
  }
  if (threadIdx.x == 0) m2[blockIdx.x] = red[0];
}

// ---------------------------------------------------------------------------
// Kernel 6: fused attention + aggregation.
// Block = 128 threads = 4 waves; wave w handles head w of dst tile blockIdx.x.
// Per 32-src chunk: build 16x32 bf16 P tile (exp of masked lrelu scores, with
// a global upper bound instead of online max), then 4x wmma_f32_16x16x32_bf16
// against V columns (D=64 -> 4 groups of 16).
// ---------------------------------------------------------------------------
__global__ void gat_attn_agg(const float* __restrict__ e1,
                             const float* __restrict__ e2,
                             const unsigned* __restrict__ bits,
                             const __bf16* __restrict__ hbT,
                             const float* __restrict__ m2,
                             float* __restrict__ out) {
  int hd   = threadIdx.x >> 5;          // wave id = head
  int lane = threadIdx.x & 31;
  int n0   = blockIdx.x * 16;           // dst tile base
  int m    = lane & 15;
  int half = lane >> 4;
  int off  = half * 8;                  // lane's K sub-offset in A tile

  float e1v   = e1[hd * N_NODES + n0 + m];
  float bt    = e1v + m2[hd];
  float bound = fmaxf(bt, ALPHA * bt);  // lrelu of upper bound

  const float*  e2h   = e2 + hd * N_NODES;
  const unsigned* brow = bits + (size_t)(n0 + m) * (N_NODES / 32);
  const __bf16* vcol  = hbT + ((size_t)(hd * HD + m)) * N_NODES;  // column d0+m

  v8f acc0 = {}, acc1 = {}, acc2 = {}, acc3 = {};
  float dsum = 0.f;

  for (int i0 = 0; i0 < N_NODES; i0 += 32) {
    unsigned mask = brow[i0 >> 5];
    __builtin_prefetch(e2h + i0 + 128, 0, 0);

    // lane's 16 e2 values: k in {off..off+7} U {16+off..16+off+7}
    float ev[16];
    ((float4*)ev)[0] = *(const float4*)(e2h + i0 + off);
    ((float4*)ev)[1] = *(const float4*)(e2h + i0 + off + 4);
    ((float4*)ev)[2] = *(const float4*)(e2h + i0 + off + 16);
    ((float4*)ev)[3] = *(const float4*)(e2h + i0 + off + 20);

    v16bf A;
#pragma unroll
    for (int j = 0; j < 16; ++j) {
      int k   = (j < 8) ? (off + j) : (8 + off + j);  // j>=8 -> 16+off+(j-8)
      float t = e1v + ev[j];
      float s = fmaxf(t, ALPHA * t);                  // leaky relu
      float w = ((mask >> k) & 1u) ? __expf(s - bound) : 0.f;
      dsum += w;
      A[j] = (__bf16)w;
    }

    // B tiles: 32x16 bf16, lane column = d0 + m, K = {half*8 + 0..7} U {16 + half*8 + 0..7}
    const __bf16* vp = vcol + i0 + half * 8;
    v16bf B0, B1, B2, B3;
    ((float4*)&B0)[0] = *(const float4*)(vp);
    ((float4*)&B0)[1] = *(const float4*)(vp + 16);
    ((float4*)&B1)[0] = *(const float4*)(vp + 16 * N_NODES);
    ((float4*)&B1)[1] = *(const float4*)(vp + 16 * N_NODES + 16);
    ((float4*)&B2)[0] = *(const float4*)(vp + 32 * N_NODES);
    ((float4*)&B2)[1] = *(const float4*)(vp + 32 * N_NODES + 16);
    ((float4*)&B3)[0] = *(const float4*)(vp + 48 * N_NODES);
    ((float4*)&B3)[1] = *(const float4*)(vp + 48 * N_NODES + 16);

    acc0 = __builtin_amdgcn_wmma_f32_16x16x32_bf16(false, A, false, B0, (short)0, acc0, false, false);
    acc1 = __builtin_amdgcn_wmma_f32_16x16x32_bf16(false, A, false, B1, (short)0, acc1, false, false);
    acc2 = __builtin_amdgcn_wmma_f32_16x16x32_bf16(false, A, false, B2, (short)0, acc2, false, false);
    acc3 = __builtin_amdgcn_wmma_f32_16x16x32_bf16(false, A, false, B3, (short)0, acc3, false, false);
  }

  // combine half-wave partial row sums: lanes m and m+16 both hold row m total
  dsum += __shfl_xor(dsum, 16, 32);
  float rcp = dsum > 0.f ? 1.0f / dsum : 0.f;

  // C layout: VGPR r -> row r + half*8, col = lane&15; normalize and store
#pragma unroll
  for (int r = 0; r < 8; ++r) {
    float sc = __shfl(rcp, r + half * 8, 32);  // lanes 0..15 hold rows 0..15
    float* op = out + (size_t)(n0 + half * 8 + r) * (NH * HD) + hd * HD + m;
    op[0]  = acc0[r] * sc;
    op[16] = acc1[r] * sc;
    op[32] = acc2[r] * sc;
    op[48] = acc3[r] * sc;
  }
}

// ---------------------------------------------------------------------------
extern "C" void kernel_launch(void* const* d_in, const int* in_sizes, int n_in,
                              void* d_out, int out_size, void* d_ws, size_t ws_size,
                              hipStream_t stream) {
  const float* x       = (const float*)d_in[0];
  const int*   adj     = (const int*)d_in[1];
  const float* W       = (const float*)d_in[2];
  const float* att_src = (const float*)d_in[3];
  const float* att_dst = (const float*)d_in[4];
  float*       out     = (float*)d_out;

  char* ws = (char*)d_ws;
  float*    h    = (float*)ws;                       // 4,194,304 B
  float*    e1   = (float*)(ws + 4194304);           //    65,536 B
  float*    e2   = (float*)(ws + 4259840);           //    65,536 B
  float*    m2   = (float*)(ws + 4325376);           //       256 B
  __bf16*   hbT  = (__bf16*)(ws + 4325632);          // 2,097,152 B
  unsigned* bits = (unsigned*)(ws + 6422784);        // 2,097,152 B  (total ~8.5MB)

  gat_gemm_h      <<<512,   256, 0, stream>>>(x, W, h);
  gat_scores      <<<64,    256, 0, stream>>>(h, att_src, att_dst, e1, e2);
  gat_transpose_bf16<<<4096, 256, 0, stream>>>(h, hbT);
  gat_pack_adj    <<<65536, 256, 0, stream>>>(adj, bits);
  gat_max_e2      <<<NH,    256, 0, stream>>>(e2, m2);
  gat_attn_agg    <<<N_NODES / 16, 128, 0, stream>>>(e1, e2, bits, hbT, m2, out);
}
